// GraphEncoder_7962869366790
// MI455X (gfx1250) — compile-verified, matrix operand order checked by reading, not run on hardware
//
#include <hip/hip_runtime.h>

// ---------- types for WMMA ----------
typedef __attribute__((ext_vector_type(16))) __bf16       v16bf;
typedef __attribute__((ext_vector_type(8)))  float        v8f;
typedef __attribute__((ext_vector_type(4)))  unsigned int v4u;

union FragBF { v16bf bf; v4u u[2]; };

__device__ __forceinline__ unsigned short f2bf(float f) {
    unsigned int u = __float_as_uint(f);
    u += 0x7FFFu + ((u >> 16) & 1u);           // round-to-nearest-even
    return (unsigned short)(u >> 16);
}

__device__ __forceinline__ void atomicMaxF(float* a, float v) {
    unsigned int old = __float_as_uint(*a);
    while (__uint_as_float(old) < v) {
        unsigned int assumed = old;
        old = atomicCAS((unsigned int*)a, assumed, __float_as_uint(v));
        if (old == assumed) break;
    }
}

// ---------- elementwise helpers ----------
__global__ void fill_f32(float* p, long n, float v) {
    long i = (long)blockIdx.x * blockDim.x + threadIdx.x;
    if (i < n) p[i] = v;
}

__global__ void cvt_bf16(const float* __restrict__ in, unsigned short* __restrict__ out, long n) {
    long i = (long)blockIdx.x * blockDim.x + threadIdx.x;
    if (i < n) out[i] = f2bf(in[i]);
}

// W is [K=256, N=256] row-major fp32 -> WT bf16 [n*256 + k]
__global__ void cvt_bf16_transpose(const float* __restrict__ W, unsigned short* __restrict__ WT) {
    int k = blockIdx.x;            // 0..255
    int n = threadIdx.x;           // 0..255
    WT[n * 256 + k] = f2bf(W[k * 256 + n]);
}

// acc[N,256] + bias -> elu -> bf16 out
__global__ void elu_bias_bf16(const float* __restrict__ acc, const float* __restrict__ b,
                              unsigned short* __restrict__ out, long n) {
    long i = (long)blockIdx.x * blockDim.x + threadIdx.x;
    if (i >= n) return;
    float v = acc[i] + b[i & 255];
    v = v > 0.f ? v : (__expf(v) - 1.f);
    out[i] = f2bf(v);
}

__global__ void add_bias(float* __restrict__ out, const float* __restrict__ b, long n) {
    long i = (long)blockIdx.x * blockDim.x + threadIdx.x;
    if (i < n) out[i] += b[i & 255];
}

// ---------- WMMA GEMM: C[M,256] = A[M,256] (bf16) x W[256,256], W given transposed bf16 ----------
// one wave computes one 16x16 tile, 8 k-steps of v_wmma_f32_16x16x32_bf16
__global__ void gemm_bf16_wmma(const unsigned short* __restrict__ A,   // [M,256] bf16 row-major
                               const unsigned short* __restrict__ BT,  // [256,256] bf16, BT[n*256+k]
                               float* __restrict__ C, int M) {
    const int K = 256, Nn = 256;
    int wave   = threadIdx.x >> 5;
    int lane   = threadIdx.x & 31;
    int tile_m = blockIdx.x * 8 + wave;
    int tile_n = blockIdx.y;
    if (tile_m * 16 >= M) return;                      // wave-uniform

    int half = lane >> 4;                              // 0: K 0-7/16-23, 1: K 8-15/24-31
    int row  = tile_m * 16 + (lane & 15);
    int col  = tile_n * 16 + (lane & 15);
    const unsigned short* arow = A  + (size_t)row * K;
    const unsigned short* brow = BT + (size_t)col * K;
    int koff = half ? 8 : 0;

    v8f c = {};
#pragma unroll
    for (int kb = 0; kb < 8; ++kb) {
        int k0 = kb * 32 + koff;
        FragBF fa, fb;
        fa.u[0] = *(const v4u*)(arow + k0);
        fa.u[1] = *(const v4u*)(arow + k0 + 16);
        fb.u[0] = *(const v4u*)(brow + k0);
        fb.u[1] = *(const v4u*)(brow + k0 + 16);
        c = __builtin_amdgcn_wmma_f32_16x16x32_bf16(false, fa.bf, false, fb.bf,
                                                    (short)0, c, false, false);
    }
    // C/D layout: VGPR v -> M = tile_m*16 + half*8 + v, N = col
    float* crow = C + (size_t)(tile_m * 16 + half * 8) * Nn + col;
#pragma unroll
    for (int v = 0; v < 8; ++v) crow[(size_t)v * Nn] = c[v];
}

// ---------- edge pass A: logits + segment max.  wave per (edge, head) ----------
__global__ void edge_logits_max(const float* __restrict__ xl, const float* __restrict__ xr,
                                const float* __restrict__ att, const int* __restrict__ ei,
                                float* __restrict__ logits, float* __restrict__ mx,
                                int E, long total, int H, int C) {
    long gw = (long)blockIdx.x * (blockDim.x >> 5) + (threadIdx.x >> 5);
    if (gw >= total) return;                            // wave-uniform
    int  lane = threadIdx.x & 31;
    long e = gw / H;
    int  h = (int)(gw - e * H);
    int  src = (e < E) ? ei[e]     : (int)(e - E);
    int  dst = (e < E) ? ei[E + e] : (int)(e - E);
    const float* pl = xl + (size_t)src * (H * C) + h * C;
    const float* pr = xr + (size_t)dst * (H * C) + h * C;
    const float* pa = att + h * C;
    float s = 0.f;
    for (int i = lane; i < C; i += 32) {
        float m = pl[i] + pr[i];
        m = m > 0.f ? m : 0.2f * m;                     // leaky_relu 0.2
        s += m * pa[i];
    }
    for (int o = 16; o > 0; o >>= 1) s += __shfl_down(s, o, 32);
    if (lane == 0) {
        logits[gw] = s;
        atomicMaxF(&mx[(size_t)dst * H + h], s);
    }
}

// ---------- edge pass B: exp + segment sum ----------
__global__ void edge_exp_denom(const int* __restrict__ ei, float* __restrict__ logits,
                               const float* __restrict__ mx, float* __restrict__ denom,
                               int E, long total, int H) {
    long idx = (long)blockIdx.x * blockDim.x + threadIdx.x;
    if (idx >= total) return;
    long e = idx / H;
    int  h = (int)(idx - e * H);
    int  dst = (e < E) ? ei[E + e] : (int)(e - E);
    float v = __expf(logits[idx] - mx[(size_t)dst * H + h]);
    logits[idx] = v;
    atomicAdd(&denom[(size_t)dst * H + h], v);
}

// ---------- edge pass C: out[dst] += alpha * xl[src].  wave per (edge, head) ----------
__global__ void edge_aggregate(const float* __restrict__ xl, const float* __restrict__ ex,
                               const float* __restrict__ denom, const int* __restrict__ ei,
                               float* __restrict__ out, int E, long total, int H, int C) {
    long gw = (long)blockIdx.x * (blockDim.x >> 5) + (threadIdx.x >> 5);
    if (gw >= total) return;
    int  lane = threadIdx.x & 31;
    long e = gw / H;
    int  h = (int)(gw - e * H);
    int  src = (e < E) ? ei[e]     : (int)(e - E);
    int  dst = (e < E) ? ei[E + e] : (int)(e - E);
    float alpha = ex[gw] / denom[(size_t)dst * H + h];
    const float* pl = xl + (size_t)src * (H * C) + h * C;
    float*       po = out + (size_t)dst * (H * C) + h * C;
    for (int i = lane; i < C; i += 32) atomicAdd(&po[i], alpha * pl[i]);
}

// ==================================================================
extern "C" void kernel_launch(void* const* d_in, const int* in_sizes, int n_in,
                              void* d_out, int out_size, void* d_ws, size_t ws_size,
                              hipStream_t stream) {
    const float* x    = (const float*)d_in[0];
    const int*   ei   = (const int*)d_in[1];
    const float* Wl1  = (const float*)d_in[2];
    const float* Wr1  = (const float*)d_in[3];
    const float* att1 = (const float*)d_in[4];
    const float* b1   = (const float*)d_in[5];
    const float* Wl2  = (const float*)d_in[6];
    const float* Wr2  = (const float*)d_in[7];
    const float* att2 = (const float*)d_in[8];
    const float* b2   = (const float*)d_in[9];
    float* out = (float*)d_out;

    const long N    = in_sizes[0] / 256;
    const int  E    = in_sizes[1] / 2;
    const long Etot = (long)E + N;

    // workspace partition
    char* p = (char*)d_ws;
    float* XL  = (float*)p; p += (size_t)N * 256 * 4;   // xl (both layers)
    float* XR  = (float*)p; p += (size_t)N * 256 * 4;   // xr (both layers)
    float* HB  = (float*)p; p += (size_t)N * 256 * 4;   // layer-1 accumulator
    unsigned short* XBF = (unsigned short*)p; p += (size_t)N * 256 * 2;
    unsigned short* WT0 = (unsigned short*)p; p += 256 * 256 * 2;
    unsigned short* WT1 = (unsigned short*)p; p += 256 * 256 * 2;
    float* LOG = (float*)p; p += (size_t)Etot * 4 * 4;  // edge logits/ex (max H=4)
    float* MX  = (float*)p; p += (size_t)N * 4 * 4;
    float* DEN = (float*)p; p += (size_t)N * 4 * 4;

    const int  T = 256;
    const long nNC = N * 256;
    const int  tiles_m = (int)((N + 15) / 16);
    dim3 gemmGrid((tiles_m + 7) / 8, 16);

    // ---------------- layer 1 (H=4, C=64) ----------------
    cvt_bf16<<<(int)((nNC + T - 1) / T), T, 0, stream>>>(x, XBF, nNC);
    cvt_bf16_transpose<<<256, 256, 0, stream>>>(Wl1, WT0);
    cvt_bf16_transpose<<<256, 256, 0, stream>>>(Wr1, WT1);
    gemm_bf16_wmma<<<gemmGrid, T, 0, stream>>>(XBF, WT0, XL, (int)N);
    gemm_bf16_wmma<<<gemmGrid, T, 0, stream>>>(XBF, WT1, XR, (int)N);

    {
        const int H = 4, C = 64;
        long totEH = Etot * H;
        fill_f32<<<(int)((N * H + T - 1) / T), T, 0, stream>>>(MX,  N * H, -3.402823466e38f);
        fill_f32<<<(int)((N * H + T - 1) / T), T, 0, stream>>>(DEN, N * H, 0.f);
        int wBlocks = (int)((totEH + 7) / 8);           // 8 waves / block
        edge_logits_max<<<wBlocks, T, 0, stream>>>(XL, XR, att1, ei, LOG, MX, E, totEH, H, C);
        edge_exp_denom<<<(int)((totEH + T - 1) / T), T, 0, stream>>>(ei, LOG, MX, DEN, E, totEH, H);
        fill_f32<<<(int)((nNC + T - 1) / T), T, 0, stream>>>(HB, nNC, 0.f);
        edge_aggregate<<<wBlocks, T, 0, stream>>>(XL, LOG, DEN, ei, HB, E, totEH, H, C);
    }
    // h = elu(acc + b1), straight to bf16 for the next GEMMs
    elu_bias_bf16<<<(int)((nNC + T - 1) / T), T, 0, stream>>>(HB, b1, XBF, nNC);

    // ---------------- layer 2 (H=1, C=256) ----------------
    cvt_bf16_transpose<<<256, 256, 0, stream>>>(Wl2, WT0);
    cvt_bf16_transpose<<<256, 256, 0, stream>>>(Wr2, WT1);
    gemm_bf16_wmma<<<gemmGrid, T, 0, stream>>>(XBF, WT0, XL, (int)N);
    gemm_bf16_wmma<<<gemmGrid, T, 0, stream>>>(XBF, WT1, XR, (int)N);

    {
        const int H = 1, C = 256;
        long totEH = Etot;
        fill_f32<<<(int)((N + T - 1) / T), T, 0, stream>>>(MX,  N, -3.402823466e38f);
        fill_f32<<<(int)((N + T - 1) / T), T, 0, stream>>>(DEN, N, 0.f);
        int wBlocks = (int)((totEH + 7) / 8);
        edge_logits_max<<<wBlocks, T, 0, stream>>>(XL, XR, att2, ei, LOG, MX, E, totEH, H, C);
        edge_exp_denom<<<(int)((totEH + T - 1) / T), T, 0, stream>>>(ei, LOG, MX, DEN, E, totEH, H);
        fill_f32<<<(int)((nNC + T - 1) / T), T, 0, stream>>>(out, nNC, 0.f);
        edge_aggregate<<<wBlocks, T, 0, stream>>>(XL, LOG, DEN, ei, out, E, totEH, H, C);
        add_bias<<<(int)((nNC + T - 1) / T), T, 0, stream>>>(out, b2, nNC);
    }
}
